// GQA_57226144252264
// MI455X (gfx1250) — compile-verified, matrix operand order checked by reading, not run on hardware
//
#include <hip/hip_runtime.h>
#include <hip/hip_bf16.h>

// ---------------------------------------------------------------------------
// GQA block for MI455X (gfx1250, wave32).  All matmuls run through
// v_wmma_f32_16x16x32_bf16 (fp32 accumulate).  Compute-bound workload
// (~150 GFLOP vs ~120MB traffic) => push everything through the WMMA pipe.
// Weights and V are pre-transposed once so every LDS tile fill in the hot
// loops is a contiguous b128 copy via GLOBAL_LOAD_ASYNC_TO_LDS_B128.
// ---------------------------------------------------------------------------

typedef __bf16 v16bf __attribute__((ext_vector_type(16)));
typedef float  v8f   __attribute__((ext_vector_type(8)));

__device__ inline v8f zero8() {
    v8f z = {0.f, 0.f, 0.f, 0.f, 0.f, 0.f, 0.f, 0.f};
    return z;
}

__device__ inline v8f wmma_bf16(v16bf a, v16bf b, v8f c) {
    // 8 args: (neg_a, A, neg_b, B, c_mod, C, reuse_a, reuse_b)
    return __builtin_amdgcn_wmma_f32_16x16x32_bf16(false, a, false, b,
                                                   (short)0, c, false, false);
}

// --- CDNA5 async global->LDS path ------------------------------------------
#if defined(__has_builtin)
#if __has_builtin(__builtin_amdgcn_global_load_async_to_lds_b128) && \
    __has_builtin(__builtin_amdgcn_s_wait_asynccnt)
#define USE_ASYNC_COPY 1
#endif
#endif

#ifdef USE_ASYNC_COPY
// Builtin prototype (from compiler diagnostic):
//   void __builtin_amdgcn_global_load_async_to_lds_b128(
//       int __vector(4) addrspace(1)* src, int __vector(4) addrspace(3)* dst,
//       imm int offset, imm int cpol);
typedef int v4i_raw __attribute__((vector_size(16)));
typedef __attribute__((address_space(1))) v4i_raw gv4i;
typedef __attribute__((address_space(3))) v4i_raw lv4i;
// AS1 pointer = full 64-bit address; AS3 pointer = low 32 bits of the generic
// LDS address (aperture maps LDS via addr[31:0]).
#define GPTR(p) ((gv4i*)(unsigned long long)(const void*)(p))
#define LPTR(p) ((lv4i*)(unsigned int)(unsigned long long)(const void*)(p))
// The same immediate offset is applied to both global and LDS addresses.
#define ASYNC_B128(g, l, off) \
    __builtin_amdgcn_global_load_async_to_lds_b128(GPTR(g), LPTR(l), (off), 0)
#define ASYNC_WAIT() __builtin_amdgcn_s_wait_asynccnt(0)
#endif

// ---------------------------------------------------------------------------
// fp32 -> bf16 convert (plain)
// ---------------------------------------------------------------------------
__global__ void cvt_f32_bf16(const float* __restrict__ in,
                             __bf16* __restrict__ out, int n) {
    int i = blockIdx.x * blockDim.x + threadIdx.x;
    if (i < n) out[i] = (__bf16)in[i];
}

// ---------------------------------------------------------------------------
// fp32 -> bf16 convert + transpose:  in[K][N] (row-major) -> out[N][K].
// One-time pass per weight matrix; removes all transpose work from the GEMM
// inner loop.
// ---------------------------------------------------------------------------
__global__ void cvt_transpose_bf16(const float* __restrict__ in,
                                   __bf16* __restrict__ out, int K, int N) {
    int i = blockIdx.x * blockDim.x + threadIdx.x;
    if (i >= K * N) return;
    int k = i / N, n = i % N;
    out[(size_t)n * K + k] = (__bf16)in[i];
}

// ---------------------------------------------------------------------------
// Tiled bf16 GEMM:  C[M,N] (fp32) = A[M,K] @ B^T, A row-major, Bt is [N][K].
// Block tile 128x128, BK=32.  256 threads = 8 waves in a 4x2 grid; each wave
// computes a 32x64 sub-tile as 2x4 WMMA 16x16 tiles (8 wmma per K-step).
// Both LDS tiles are K-contiguous, so fragment reads are 32B ds_load_b128
// pairs and tile fills are pure b128 async-to-LDS copies.
// ---------------------------------------------------------------------------
#define GTM 128
#define GTN 128
#define GTK 32

__global__ void __launch_bounds__(256)
gemm_bf16_kernel(const __bf16* __restrict__ A, const __bf16* __restrict__ Bt,
                 float* __restrict__ C, int Mdim, int Ndim, int Kdim) {
    (void)Mdim;
    __shared__ alignas(64) __bf16 lds_a[GTM * GTK];  // [m][k]
    __shared__ alignas(64) __bf16 lds_b[GTN * GTK];  // [n][k]

    const int t     = threadIdx.x;
    const int lane  = t & 31;
    const int wid   = t >> 5;
    const int wm    = wid >> 1;  // 0..3  (rows of 32)
    const int wn    = wid & 1;   // 0..1  (cols of 64)
    const int arow  = lane & 15;
    const int ahalf = lane >> 4;
    const int bm    = blockIdx.y, bn = blockIdx.x;

    v8f acc[2][4];
#pragma unroll
    for (int i = 0; i < 2; i++)
#pragma unroll
        for (int j = 0; j < 4; j++) acc[i][j] = zero8();

    const int f_row = t >> 1, f_seg = (t & 1) * 16;  // 128 rows x 2 segments

    const __bf16* srcA = A + (size_t)(bm * GTM + f_row) * Kdim + f_seg;
    const __bf16* srcB = Bt + (size_t)(bn * GTN + f_row) * Kdim + f_seg;
    __bf16* dstA = lds_a + f_row * GTK + f_seg;
    __bf16* dstB = lds_b + f_row * GTK + f_seg;

    for (int k0 = 0; k0 < Kdim; k0 += GTK) {
        __syncthreads();
#ifdef USE_ASYNC_COPY
        ASYNC_B128(srcA + k0, dstA, 0);
        ASYNC_B128(srcA + k0, dstA, 16);
        ASYNC_B128(srcB + k0, dstB, 0);
        ASYNC_B128(srcB + k0, dstB, 16);
#else
        *(v16bf*)dstA = *(const v16bf*)(srcA + k0);
        *(v16bf*)dstB = *(const v16bf*)(srcB + k0);
#endif
        if (k0 + GTK < Kdim) {
            __builtin_prefetch(srcA + k0 + GTK, 0, 1);
            __builtin_prefetch(srcB + k0 + GTK, 0, 1);
        }
#ifdef USE_ASYNC_COPY
        ASYNC_WAIT();
#endif
        __syncthreads();

        v16bf af[2], bf[4];
#pragma unroll
        for (int i = 0; i < 2; i++)
            af[i] = *(const v16bf*)(lds_a + (wm * 32 + i * 16 + arow) * GTK +
                                    ahalf * 16);
#pragma unroll
        for (int j = 0; j < 4; j++)
            bf[j] = *(const v16bf*)(lds_b + (wn * 64 + j * 16 + arow) * GTK +
                                    ahalf * 16);
#pragma unroll
        for (int i = 0; i < 2; i++)
#pragma unroll
            for (int j = 0; j < 4; j++)
                acc[i][j] = wmma_bf16(af[i], bf[j], acc[i][j]);
    }

    // Epilogue: C-layout lane (VGPR r, lane l) -> (m = r + 8*(l>=16), n = l%16)
#pragma unroll
    for (int i = 0; i < 2; i++)
#pragma unroll
        for (int j = 0; j < 4; j++)
#pragma unroll
            for (int r = 0; r < 8; r++) {
                int row = bm * GTM + wm * 32 + i * 16 + r + ahalf * 8;
                int col = bn * GTN + wn * 64 + j * 16 + arow;
                C[(size_t)row * Ndim + col] = acc[i][j][r];
            }
}

// ---------------------------------------------------------------------------
// RMSNorm + RoPE.  One wave per (b, s, head) row of 128 elements.
// Source layout: proj[(b*S+s)*srcld + d*nheads + h]   (reference reshape)
// Dest layout  : out[((b*nheads+h)*S + s)*128 + d]    (head-major, bf16)
// ---------------------------------------------------------------------------
__global__ void __launch_bounds__(256)
normrope_kernel(const float* __restrict__ proj, const float* __restrict__ rms_w,
                const int* __restrict__ pos, __bf16* __restrict__ out,
                int nheads, int srcld, int S_) {
    const int wid  = (blockIdx.x * blockDim.x + threadIdx.x) >> 5;
    const int lane = threadIdx.x & 31;
    const int h = wid % nheads;
    const int s = (wid / nheads) % S_;
    const int b = wid / (nheads * S_);

    const float* src = proj + (size_t)(b * S_ + s) * srcld;
    float v[4];
#pragma unroll
    for (int i = 0; i < 4; i++) v[i] = src[(lane * 4 + i) * nheads + h];

    float ss = v[0] * v[0] + v[1] * v[1] + v[2] * v[2] + v[3] * v[3];
#pragma unroll
    for (int o = 16; o >= 1; o >>= 1) ss += __shfl_xor(ss, o, 32);
    const float r = rsqrtf(ss * (1.0f / 128.0f) + 1e-6f);
#pragma unroll
    for (int i = 0; i < 4; i++) v[i] = v[i] * r * rms_w[lane * 4 + i];

    const float p = (float)pos[s];
    __bf16* dst = out + ((size_t)(b * nheads + h) * S_ + s) * 128 + lane * 4;
#pragma unroll
    for (int jj = 0; jj < 2; jj++) {
        const int j = lane * 2 + jj;  // rotary pair index
        float invf = __powf(10000.0f, -(float)(2 * j) * (1.0f / 128.0f));
        float sn, cs;
        __sincosf(p * invf, &sn, &cs);
        float x1 = v[jj * 2], x2 = v[jj * 2 + 1];
        dst[jj * 2]     = (__bf16)(x1 * cs - x2 * sn);
        dst[jj * 2 + 1] = (__bf16)(x1 * sn + x2 * cs);
    }
}

// ---------------------------------------------------------------------------
// V reorder into d-major layout so attention V tiles are contiguous:
// vf[(b*S+s)*(128*nkv) + d*nkv + kv]  ->  vbh[((b*nkv+kv)*128 + d)*S + s]
// ---------------------------------------------------------------------------
__global__ void reorder_v_kernel(const float* __restrict__ vf,
                                 __bf16* __restrict__ out, int S_, int nkv,
                                 int n) {
    int i = blockIdx.x * blockDim.x + threadIdx.x;
    if (i >= n) return;
    const int cols = 128 * nkv;
    int col = i % cols, row = i / cols;
    int d = col / nkv, kv = col % nkv;
    int s = row % S_, b = row / S_;
    out[((size_t)(b * nkv + kv) * 128 + d) * S_ + s] = (__bf16)vf[i];
}

// ---------------------------------------------------------------------------
// Causal flash attention, HD=128.  Grid: (S/64, NH, B); 128 threads = 4 waves.
// Wave w owns query rows [q0 + 16w, q0 + 16w + 16).  Keys processed in chunks
// of 32: 8 wmma for QK^T, online softmax in the C-matrix layout, P restaged
// through LDS into one A-fragment, then 8 wmma for P@V (16x128 fp32 acc).
// K is key-major, V is d-major -> both tile fills are contiguous b128 copies.
// ---------------------------------------------------------------------------
__global__ void __launch_bounds__(128)
attn_kernel(const __bf16* __restrict__ Q, const __bf16* __restrict__ Kt,
            const __bf16* __restrict__ Vt, __bf16* __restrict__ ctx, int S_,
            int nh, int g) {
    __shared__ alignas(64) __bf16 lds_k[32 * 128];    // [key][d]
    __shared__ alignas(64) __bf16 lds_v[128 * 32];    // [d][key]
    __shared__ alignas(64) __bf16 lds_p[4][16 * 32];  // per-wave P staging

    const int lane  = threadIdx.x & 31;
    const int wave  = threadIdx.x >> 5;
    const int arow  = lane & 15;
    const int ahalf = lane >> 4;
    const int b = blockIdx.z, h = blockIdx.y;
    const int nkv = nh / g, kv = h / g;
    const int q0   = blockIdx.x * 64;
    const int qrow = q0 + wave * 16;

    const __bf16* Qh = Q + ((size_t)(b * nh + h) * S_ + qrow) * 128;
    const __bf16* Kh = Kt + ((size_t)(b * nkv + kv) * S_) * 128;
    const __bf16* Vh = Vt + ((size_t)(b * nkv + kv) * 128) * S_;  // d-major

    // Q fragments for the wave's 16 rows, K-dim in 4 chunks of 32
    v16bf qf[4];
#pragma unroll
    for (int c = 0; c < 4; c++)
        qf[c] = *(const v16bf*)(Qh + (size_t)arow * 128 + c * 32 + ahalf * 16);

    v8f acc[8];
#pragma unroll
    for (int t = 0; t < 8; t++) acc[t] = zero8();
    float rm[8], rl[8];
#pragma unroll
    for (int r = 0; r < 8; r++) { rm[r] = -1e30f; rl[r] = 0.f; }

    const int fr   = threadIdx.x >> 2;        // 0..31 (key row for K fill)
    const int fseg = (threadIdx.x & 3) * 32;  // d segment for K fill
    const int vd   = threadIdx.x;             // 0..127 (d row for V fill)
    const int nchunks = (q0 + 64) / 32;
    const float scale = 0.088388347648318447f;  // 1/sqrt(128)

    for (int ch = 0; ch < nchunks; ch++) {
        const int kb = ch * 32;
        __syncthreads();
        {
            const __bf16* srck = Kh + (size_t)(kb + fr) * 128 + fseg;
            __bf16* dstk = lds_k + fr * 128 + fseg;
            const __bf16* srcv = Vh + (size_t)vd * S_ + kb;
            __bf16* dstv = lds_v + vd * 32;
#ifdef USE_ASYNC_COPY
            ASYNC_B128(srck, dstk, 0);
            ASYNC_B128(srck, dstk, 16);
            ASYNC_B128(srck, dstk, 32);
            ASYNC_B128(srck, dstk, 48);
            ASYNC_B128(srcv, dstv, 0);
            ASYNC_B128(srcv, dstv, 16);
            ASYNC_B128(srcv, dstv, 32);
            ASYNC_B128(srcv, dstv, 48);
            ASYNC_WAIT();
#else
            *(v16bf*)dstk        = *(const v16bf*)srck;
            *(v16bf*)(dstk + 16) = *(const v16bf*)(srck + 16);
            *(v16bf*)dstv        = *(const v16bf*)srcv;
            *(v16bf*)(dstv + 16) = *(const v16bf*)(srcv + 16);
#endif
        }
        __syncthreads();

        // Scores: two 16-key tiles, K-dim 128 in 4 WMMA steps each
        v8f sc0 = zero8(), sc1 = zero8();
#pragma unroll
        for (int ks = 0; ks < 4; ks++) {
            v16bf kf0 = *(const v16bf*)(lds_k + (size_t)arow * 128 + ks * 32 +
                                        ahalf * 16);
            v16bf kf1 = *(const v16bf*)(lds_k + (size_t)(16 + arow) * 128 +
                                        ks * 32 + ahalf * 16);
            sc0 = wmma_bf16(qf[ks], kf0, sc0);
            sc1 = wmma_bf16(qf[ks], kf1, sc1);
        }

        // Online softmax (row stats live replicated across each 16-lane half)
        const int kc0 = kb + arow, kc1 = kb + 16 + arow;
#pragma unroll
        for (int r = 0; r < 8; r++) {
            const int m = r + ahalf * 8;
            const int qpos = qrow + m;
            float s0 = (kc0 <= qpos) ? sc0[r] * scale : -1e30f;
            float s1 = (kc1 <= qpos) ? sc1[r] * scale : -1e30f;
            float mx = fmaxf(s0, s1);
#pragma unroll
            for (int o = 8; o >= 1; o >>= 1)
                mx = fmaxf(mx, __shfl_xor(mx, o, 32));
            const float mnew = fmaxf(rm[r], mx);
            const float corr = __expf(rm[r] - mnew);
            const float p0 = __expf(s0 - mnew);
            const float p1 = __expf(s1 - mnew);
            float ls = p0 + p1;
#pragma unroll
            for (int o = 8; o >= 1; o >>= 1) ls += __shfl_xor(ls, o, 32);
            rl[r] = rl[r] * corr + ls;
            rm[r] = mnew;
#pragma unroll
            for (int t = 0; t < 8; t++) acc[t][r] *= corr;
            lds_p[wave][m * 32 + arow]      = (__bf16)p0;
            lds_p[wave][m * 32 + 16 + arow] = (__bf16)p1;
        }
        __syncthreads();

        // P (16x32) is exactly one A-fragment; P@V over 8 d-tiles
        v16bf pf = *(const v16bf*)(&lds_p[wave][arow * 32 + ahalf * 16]);
#pragma unroll
        for (int t = 0; t < 8; t++) {
            v16bf vf2 =
                *(const v16bf*)(lds_v + (t * 16 + arow) * 32 + ahalf * 16);
            acc[t] = wmma_bf16(pf, vf2, acc[t]);
        }
    }

    // Normalize and emit ctx in [b*S+s, h*128+d] row-major bf16
    const int Hld = nh * 128;
#pragma unroll
    for (int t = 0; t < 8; t++)
#pragma unroll
        for (int r = 0; r < 8; r++) {
            const int m = r + ahalf * 8;
            float o = acc[t][r] / rl[r];
            ctx[((size_t)(b * S_) + qrow + m) * Hld + h * 128 + t * 16 + arow] =
                (__bf16)o;
        }
}

// ---------------------------------------------------------------------------
// Host launcher
// ---------------------------------------------------------------------------
extern "C" void kernel_launch(void* const* d_in, const int* in_sizes, int n_in,
                              void* d_out, int out_size, void* d_ws,
                              size_t ws_size, hipStream_t stream) {
    (void)in_sizes; (void)n_in; (void)out_size; (void)ws_size;
    const float* x     = (const float*)d_in[0];
    const float* wq    = (const float*)d_in[1];
    const float* wk    = (const float*)d_in[2];
    const float* wv    = (const float*)d_in[3];
    const float* wo    = (const float*)d_in[4];
    const float* rms_w = (const float*)d_in[5];
    const int*   pos   = (const int*)d_in[6];

    enum { Bn = 2, Sn = 2048, Hn = 2048, NHn = 16, Gn = 2, NKVn = 8 };
    const int M  = Bn * Sn;  // 4096
    const int Hk = Hn / Gn;  // 1024

    char* base = (char*)d_ws;
    size_t off = 0;
    auto take = [&](size_t bytes) -> char* {
        char* p = base + off;
        off = (off + bytes + 255) & ~(size_t)255;
        return p;
    };
    __bf16* xb   = (__bf16*)take((size_t)M * Hn * 2);
    __bf16* wqb  = (__bf16*)take((size_t)Hn * Hn * 2);  // transposed [N][K]
    __bf16* wkb  = (__bf16*)take((size_t)Hn * Hk * 2);  // transposed
    __bf16* wvb  = (__bf16*)take((size_t)Hn * Hk * 2);  // transposed
    __bf16* wob  = (__bf16*)take((size_t)Hn * Hn * 2);  // transposed
    float*  qf   = (float*)take((size_t)M * Hn * 4);
    float*  kf   = (float*)take((size_t)M * Hk * 4);
    float*  vf   = (float*)take((size_t)M * Hk * 4);
    __bf16* qbh  = (__bf16*)take((size_t)M * Hn * 2);
    __bf16* kbh  = (__bf16*)take((size_t)M * Hk * 2);
    __bf16* vbh  = (__bf16*)take((size_t)M * Hk * 2);
    __bf16* ctxb = (__bf16*)qf;  // qf is dead after normrope_q; reuse

    const int thr = 256;
    cvt_f32_bf16<<<(M * Hn + thr - 1) / thr, thr, 0, stream>>>(x, xb, M * Hn);
    cvt_transpose_bf16<<<(Hn * Hn + thr - 1) / thr, thr, 0, stream>>>(wq, wqb,
                                                                      Hn, Hn);
    cvt_transpose_bf16<<<(Hn * Hk + thr - 1) / thr, thr, 0, stream>>>(wk, wkb,
                                                                      Hn, Hk);
    cvt_transpose_bf16<<<(Hn * Hk + thr - 1) / thr, thr, 0, stream>>>(wv, wvb,
                                                                      Hn, Hk);
    cvt_transpose_bf16<<<(Hn * Hn + thr - 1) / thr, thr, 0, stream>>>(wo, wob,
                                                                      Hn, Hn);

    dim3 gq(Hn / GTN, M / GTM);
    gemm_bf16_kernel<<<gq, 256, 0, stream>>>(xb, wqb, qf, M, Hn, Hn);
    dim3 gkv(Hk / GTN, M / GTM);
    gemm_bf16_kernel<<<gkv, 256, 0, stream>>>(xb, wkb, kf, M, Hk, Hn);
    gemm_bf16_kernel<<<gkv, 256, 0, stream>>>(xb, wvb, vf, M, Hk, Hn);

    normrope_kernel<<<(M * NHn) / 8, 256, 0, stream>>>(qf, rms_w, pos, qbh,
                                                       NHn, Hn, Sn);
    normrope_kernel<<<(M * NKVn) / 8, 256, 0, stream>>>(kf, rms_w, pos, kbh,
                                                        NKVn, Hk, Sn);
    reorder_v_kernel<<<(M * Hk + thr - 1) / thr, thr, 0, stream>>>(
        vf, vbh, Sn, NKVn, M * Hk);

    dim3 ga(Sn / 64, NHn, Bn);
    attn_kernel<<<ga, 128, 0, stream>>>(qbh, kbh, vbh, ctxb, Sn, NHn, Gn);

    dim3 go(Hn / GTN, M / GTM);
    gemm_bf16_kernel<<<go, 256, 0, stream>>>(ctxb, wob, (float*)d_out, M, Hn,
                                             Hn);
}
